// MotionFormerDecoderLayer_2070174236876
// MI455X (gfx1250) — matrix operand order repaired and checked by reading, not compiled, not run on hardware
//
#include <hip/hip_runtime.h>
#include <hip/hip_bf16.h>
#include <cstdint>

// ---------------- problem constants (fixed by reference) ----------------
constexpr int B_ = 8, Q_ = 1024, S_ = 4096, D_ = 192, NH_ = 6, NP_ = 4;
constexpr int DFF_ = 768, HF_ = 64, WF_ = 64, DH_ = 32; // DH = D/NH

typedef _Float16 half16 __attribute__((ext_vector_type(16)));
typedef _Float16 half8  __attribute__((ext_vector_type(8)));
typedef float    floatx8 __attribute__((ext_vector_type(8)));

union H16 { half16 v; half8 h[2]; };

__device__ __forceinline__ floatx8 zero8() {
  floatx8 z;
#pragma unroll
  for (int i = 0; i < 8; ++i) z[i] = 0.0f;
  return z;
}

__device__ __forceinline__ float gelu_exact(float x) {
  return 0.5f * x * (1.0f + erff(x * 0.70710678118654752f));
}

// ---------------- generic conversion kernels ----------------
__global__ void cvt_f32_to_f16_kernel(const float* __restrict__ in,
                                      _Float16* __restrict__ out, size_t n) {
  size_t i = (size_t)blockIdx.x * blockDim.x + threadIdx.x;
  size_t stride = (size_t)gridDim.x * blockDim.x;
  for (; i < n; i += stride) out[i] = (_Float16)in[i];
}

__global__ void add_cvt_f16_kernel(const float* __restrict__ a,
                                   const float* __restrict__ b,
                                   _Float16* __restrict__ out, size_t n) {
  size_t i = (size_t)blockIdx.x * blockDim.x + threadIdx.x;
  size_t stride = (size_t)gridDim.x * blockDim.x;
  for (; i < n; i += stride) out[i] = (_Float16)(a[i] + b[i]);
}

// ---------------- WMMA GEMM:  C[M,N] = A[M,K] @ W[N,K]^T + bias ----------------
// A row-major f16 (K contiguous), W row-major f16 (K contiguous, PyTorch (out,in)).
// One wave computes a 16 x (NT*16) tile; the A fragment is reused across NT B
// fragments.  The wave id is forced into an SGPR via readfirstlane so that all
// tile guards compile to scalar branches (EXEC stays all-ones across WMMAs).
// A-operand lane layout (16x32 f16): row = lane&15, k = (lane>>4)*8 + (h<8 ? h : 16+h-8)
// B-operand lane layout (32x16 f16): col = lane&15, k = (lane>>4)*8 + (h<8 ? h : 16+h-8)
// act: 0 = none, 1 = exact GELU, 2 = scale by 1/sqrt(DH) (folded softmax scale)
template <int NT>
__global__ __launch_bounds__(256)
void wmma_gemm_kernel(const _Float16* __restrict__ A,
                      const _Float16* __restrict__ W,
                      const float* __restrict__ bias,
                      float* __restrict__ outF,
                      _Float16* __restrict__ outH,
                      int M, int N, int K, int act) {
  const int lane = threadIdx.x & 31;
  const int wave = __builtin_amdgcn_readfirstlane(
      blockIdx.x * (int)(blockDim.x >> 5) + (int)(threadIdx.x >> 5));
  const int mt = M >> 4;
  const int nt = (N + NT * 16 - 1) / (NT * 16);
  if (wave >= mt * nt) return;                 // scalar branch
  const int m0 = (wave / nt) << 4;
  const int n0 = (wave % nt) * (NT * 16);
  const int col = lane & 15;
  const int hi = lane >> 4;
  const int abase = hi * 8;
  const int bko = hi * 8;

  floatx8 c[NT];
#pragma unroll
  for (int t = 0; t < NT; ++t) c[t] = zero8();

  // per-subtile weight row pointers (clamped for the ragged N==24 case)
  const _Float16* Wr[NT];
#pragma unroll
  for (int t = 0; t < NT; ++t) {
    int wr = n0 + t * 16 + col;
    if (wr > N - 1) wr = N - 1;                // per-lane select, no branch
    Wr[t] = W + (size_t)wr * K + bko;
  }

  const _Float16* Ar = A + (size_t)(m0 + col) * K + abase;
  for (int k = 0; k < K; k += 32) {            // scalar loop
    H16 a;
    a.h[0] = *(const half8*)(Ar + k);
    a.h[1] = *(const half8*)(Ar + k + 16);
#pragma unroll
    for (int t = 0; t < NT; ++t) {
      H16 bfrag;
      bfrag.h[0] = *(const half8*)(Wr[t] + k);
      bfrag.h[1] = *(const half8*)(Wr[t] + k + 16);
      c[t] = __builtin_amdgcn_wmma_f32_16x16x32_f16(
          false, a.v, false, bfrag.v, (short)0, c[t], false, false);
    }
  }

#pragma unroll
  for (int t = 0; t < NT; ++t) {
    int gn = n0 + t * 16 + col;
    if (gn < N) {                              // divergent only after all WMMAs
      float bv = bias ? bias[gn] : 0.0f;
#pragma unroll
      for (int r = 0; r < 8; ++r) {
        int m = m0 + r + hi * 8;
        float v = c[t][r] + bv;
        if (act == 1) v = gelu_exact(v);
        else if (act == 2) v *= 0.17677669529663688f;   // 1/sqrt(32)
        size_t oi = (size_t)m * N + gn;
        if (outF) outF[oi] = v;
        if (outH) outH[oi] = (_Float16)v;
      }
    }
  }
}

// ---------------- V transpose: (B,Q,D) f16 -> (B,NH,DH,Q) f16 ----------------
__global__ void vtrans_kernel(const _Float16* __restrict__ vh,
                              _Float16* __restrict__ vT) {
  size_t t = (size_t)blockIdx.x * blockDim.x + threadIdx.x;
  const size_t total = (size_t)B_ * Q_ * D_;
  if (t >= total) return;
  int dfull = (int)(t % D_);
  size_t bq = t / D_;
  int q = (int)(bq % Q_);
  int b = (int)(bq / Q_);
  int h = dfull / DH_, d = dfull % DH_;
  vT[(((size_t)(b * NH_ + h)) * DH_ + d) * Q_ + q] = vh[t];
}

// ---------------- Flash self-attention, one wave per (b, h, 16-query tile) ----
// Q was pre-scaled by 1/sqrt(dh) in its projection epilogue.
// scoresT tile (keys x queries) so that two consecutive 16-key C fragments
// concatenate exactly into the A-operand layout of the P@V WMMA.
__global__ __launch_bounds__(256)
void flash_attn_kernel(const _Float16* __restrict__ qh,
                       const _Float16* __restrict__ kh,
                       const _Float16* __restrict__ vT,
                       _Float16* __restrict__ out) {
  const int lane = threadIdx.x & 31;
  const int wid = __builtin_amdgcn_readfirstlane(
      blockIdx.x * (int)(blockDim.x >> 5) + (int)(threadIdx.x >> 5));
  const int QT = Q_ / 16;
  if (wid >= B_ * NH_ * QT) return;            // scalar branch
  const int qt = wid % QT;
  const int bh = wid / QT;
  const int h = bh % NH_;
  const int b = bh / NH_;
  const int q0 = qt * 16;
  const int col = lane & 15;
  const int hi = lane >> 4;
  const int abase = hi * 8;

  // B operand: 16 queries, K=dh=32
  H16 bq;
  {
    const _Float16* qr = qh + ((size_t)(b * Q_ + q0 + col)) * D_ + h * DH_ + hi * 8;
    bq.h[0] = *(const half8*)(qr);
    bq.h[1] = *(const half8*)(qr + 16);
  }

  floatx8 acc0 = zero8(), acc1 = zero8();
  float run_max = -1e30f, run_sum = 0.0f;

  const _Float16* kbase = kh + ((size_t)b * Q_) * D_ + h * DH_;
  const _Float16* vt0 = vT + (((size_t)(b * NH_ + h)) * DH_ + col) * Q_;
  const _Float16* vt1 = vT + (((size_t)(b * NH_ + h)) * DH_ + 16 + col) * Q_;

  for (int s = 0; s < Q_; s += 32) {
    H16 a0, a1;
    const _Float16* kr0 = kbase + (size_t)(s + col) * D_;
    a0.h[0] = *(const half8*)(kr0 + abase);
    a0.h[1] = *(const half8*)(kr0 + abase + 16);
    const _Float16* kr1 = kbase + (size_t)(s + 16 + col) * D_;
    a1.h[0] = *(const half8*)(kr1 + abase);
    a1.h[1] = *(const half8*)(kr1 + abase + 16);

    floatx8 z = zero8();
    floatx8 c0 = __builtin_amdgcn_wmma_f32_16x16x32_f16(
        false, a0.v, false, bq.v, (short)0, z, false, false);
    floatx8 c1 = __builtin_amdgcn_wmma_f32_16x16x32_f16(
        false, a1.v, false, bq.v, (short)0, z, false, false);

    // online softmax over keys; lane holds query=lane&15 (replicated in lane+16)
    float mloc = -1e30f;
#pragma unroll
    for (int i = 0; i < 8; ++i) mloc = fmaxf(mloc, fmaxf(c0[i], c1[i]));
    mloc = fmaxf(mloc, __shfl_xor(mloc, 16, 32));
    float mnew = fmaxf(run_max, mloc);
    float alpha = __expf(run_max - mnew);
    run_max = mnew;

    H16 p;
    float lsum = 0.0f;
#pragma unroll
    for (int i = 0; i < 8; ++i) { float e = __expf(c0[i] - mnew); lsum += e; p.v[i] = (_Float16)e; }
#pragma unroll
    for (int i = 0; i < 8; ++i) { float e = __expf(c1[i] - mnew); lsum += e; p.v[8 + i] = (_Float16)e; }
    lsum += __shfl_xor(lsum, 16, 32);
    run_sum = run_sum * alpha + lsum;

    // rescale accumulators: acc row m = r + hi*8; alpha for query q lives in lane q
#pragma unroll
    for (int r = 0; r < 8; ++r) {
      float al = __shfl(alpha, r + abase, 32);
      acc0[r] *= al; acc1[r] *= al;
    }

    H16 bv0, bv1;
    bv0.h[0] = *(const half8*)(vt0 + s + hi * 8);
    bv0.h[1] = *(const half8*)(vt0 + s + hi * 8 + 16);
    bv1.h[0] = *(const half8*)(vt1 + s + hi * 8);
    bv1.h[1] = *(const half8*)(vt1 + s + hi * 8 + 16);
    acc0 = __builtin_amdgcn_wmma_f32_16x16x32_f16(
        false, p.v, false, bv0.v, (short)0, acc0, false, false);
    acc1 = __builtin_amdgcn_wmma_f32_16x16x32_f16(
        false, p.v, false, bv1.v, (short)0, acc1, false, false);
  }

#pragma unroll
  for (int r = 0; r < 8; ++r) {
    float sden = __shfl(run_sum, r + abase, 32);
    float inv = 1.0f / sden;
    int m = q0 + r + abase;
    _Float16* orow = out + ((size_t)(b * Q_ + m)) * D_ + h * DH_;
    orow[col] = (_Float16)(acc0[r] * inv);
    orow[16 + col] = (_Float16)(acc1[r] * inv);
  }
}

// ---------------- LayerNorm(res + y), wave per row, D=192 -> 6 elems/lane ----
__global__ __launch_bounds__(256)
void ln_kernel(const float* __restrict__ res, const float* __restrict__ yv,
               const float* __restrict__ g, const float* __restrict__ bb,
               float* __restrict__ outF, _Float16* __restrict__ outH, int rows) {
  int lane = threadIdx.x & 31;
  int row = __builtin_amdgcn_readfirstlane(
      blockIdx.x * (int)(blockDim.x >> 5) + (int)(threadIdx.x >> 5));
  if (row >= rows) return;
  const float* r0 = res + (size_t)row * D_;
  const float* y0 = yv + (size_t)row * D_;
  float v[6];
  float s = 0.0f;
#pragma unroll
  for (int j = 0; j < 6; ++j) { int e = lane + 32 * j; v[j] = r0[e] + y0[e]; s += v[j]; }
#pragma unroll
  for (int o = 16; o > 0; o >>= 1) s += __shfl_xor(s, o, 32);
  float mean = s * (1.0f / D_);
  float s2 = 0.0f;
#pragma unroll
  for (int j = 0; j < 6; ++j) { float d = v[j] - mean; s2 += d * d; }
#pragma unroll
  for (int o = 16; o > 0; o >>= 1) s2 += __shfl_xor(s2, o, 32);
  float rstd = rsqrtf(s2 * (1.0f / D_) + 1e-5f);
  float* oF = outF ? outF + (size_t)row * D_ : nullptr;
  _Float16* oH = outH ? outH + (size_t)row * D_ : nullptr;
#pragma unroll
  for (int j = 0; j < 6; ++j) {
    int e = lane + 32 * j;
    float o = (v[j] - mean) * rstd * g[e] + bb[e];
    if (oF) oF[e] = o;
    if (oH) oH[e] = (_Float16)o;
  }
}

// ---------------- deformable sampling + weighting ----------------
// thread = (b,q,h,d); reads 4 sampling points * 4 bilinear taps from vproj f32.
__global__ __launch_bounds__(256)
void deform_kernel(const float* __restrict__ vproj, const float* __restrict__ offb,
                   const float* __restrict__ awl, const float* __restrict__ refp,
                   _Float16* __restrict__ cross16) {
  int t = blockIdx.x * blockDim.x + threadIdx.x;
  const int total = B_ * Q_ * NH_ * DH_;
  if (t >= total) return;
  int d = t & (DH_ - 1);
  int rest = t / DH_;
  int h = rest % NH_;
  int bq = rest / NH_;
  int b = bq / Q_;

  const float* lg = awl + (size_t)bq * (NH_ * NP_) + h * NP_;
  float l0 = lg[0], l1 = lg[1], l2 = lg[2], l3 = lg[3];
  float mx = fmaxf(fmaxf(l0, l1), fmaxf(l2, l3));
  float e0 = __expf(l0 - mx), e1 = __expf(l1 - mx);
  float e2 = __expf(l2 - mx), e3 = __expf(l3 - mx);
  float inv = 1.0f / (e0 + e1 + e2 + e3);
  float aw[4] = {e0 * inv, e1 * inv, e2 * inv, e3 * inv};

  const float* offr = offb + (size_t)bq * (NH_ * NP_ * 2) + h * NP_ * 2;
  float rx = refp[(size_t)bq * 2 + 0];
  float ry = refp[(size_t)bq * 2 + 1];
  const float* vb = vproj + ((size_t)b * S_) * D_ + h * DH_ + d;

  float acc = 0.0f;
#pragma unroll
  for (int p = 0; p < NP_; ++p) {
    float px = (rx + offr[p * 2 + 0] * (1.0f / WF_)) * WF_ - 0.5f;
    float py = (ry + offr[p * 2 + 1] * (1.0f / HF_)) * HF_ - 0.5f;
    float x0f = floorf(px), y0f = floorf(py);
    float wx = px - x0f, wy = py - y0f;
#pragma unroll
    for (int dy = 0; dy < 2; ++dy) {
#pragma unroll
      for (int dx = 0; dx < 2; ++dx) {
        float xi = x0f + dx, yi = y0f + dy;
        float w = (dx ? wx : 1.0f - wx) * (dy ? wy : 1.0f - wy);
        bool valid = (xi >= 0.0f) && (xi <= (float)(WF_ - 1)) &&
                     (yi >= 0.0f) && (yi <= (float)(HF_ - 1));
        float xc = fminf(fmaxf(xi, 0.0f), (float)(WF_ - 1));
        float yc = fminf(fmaxf(yi, 0.0f), (float)(HF_ - 1));
        int s = (int)yc * WF_ + (int)xc;
        float val = vb[(size_t)s * D_];
        acc += aw[p] * (valid ? w : 0.0f) * val;
      }
    }
  }
  cross16[t] = (_Float16)acc;  // t == bq*D_ + h*DH_ + d
}

// ================== host-side launch sequence ==================
extern "C" void kernel_launch(void* const* d_in, const int* in_sizes, int n_in,
                              void* d_out, int out_size, void* d_ws, size_t ws_size,
                              hipStream_t stream) {
  (void)in_sizes; (void)n_in; (void)out_size; (void)ws_size;

  const float* query = (const float*)d_in[0];
  const float* pos   = (const float*)d_in[1];
  const float* src   = (const float*)d_in[2];
  const float* refp  = (const float*)d_in[3];
  const float* Wq = (const float*)d_in[4];  const float* bq = (const float*)d_in[5];
  const float* Wk = (const float*)d_in[6];  const float* bk = (const float*)d_in[7];
  const float* Wv = (const float*)d_in[8];  const float* bv = (const float*)d_in[9];
  const float* Wo = (const float*)d_in[10]; const float* bo = (const float*)d_in[11];
  const float* ln1g = (const float*)d_in[12]; const float* ln1b = (const float*)d_in[13];
  const float* dvW = (const float*)d_in[14];  const float* dvb = (const float*)d_in[15];
  const float* doffW = (const float*)d_in[16]; const float* doffb = (const float*)d_in[17];
  const float* dawW = (const float*)d_in[18];  const float* dawb = (const float*)d_in[19];
  const float* doutW = (const float*)d_in[20]; const float* doutb = (const float*)d_in[21];
  const float* ln2g = (const float*)d_in[22]; const float* ln2b = (const float*)d_in[23];
  const float* W1 = (const float*)d_in[24]; const float* b1 = (const float*)d_in[25];
  const float* W2 = (const float*)d_in[26]; const float* b2 = (const float*)d_in[27];
  const float* ln3g = (const float*)d_in[28]; const float* ln3b = (const float*)d_in[29];
  float* outp = (float*)d_out;

  const size_t BQ = (size_t)B_ * Q_;          // 8192 rows
  const size_t BS = (size_t)B_ * S_;          // 32768 rows
  const size_t nBQD = BQ * D_;
  const size_t nBSD = BS * D_;

  uint8_t* base = (uint8_t*)d_ws;
  size_t po = 0;
  auto palloc = [&](size_t bytes) -> void* {
    void* p = base + po; po = (po + bytes + 255) & ~(size_t)255; return p;
  };

  // persistent: f16 weights + cross-phase activations
  _Float16* Wq16 = ( _Float16*)palloc(D_ * D_ * 2);
  _Float16* Wk16 = ( _Float16*)palloc(D_ * D_ * 2);
  _Float16* Wv16 = ( _Float16*)palloc(D_ * D_ * 2);
  _Float16* Wo16 = ( _Float16*)palloc(D_ * D_ * 2);
  _Float16* dvW16 = ( _Float16*)palloc(D_ * D_ * 2);
  _Float16* doffW16 = ( _Float16*)palloc(NH_ * NP_ * 2 * D_ * 2);
  _Float16* dawW16 = ( _Float16*)palloc(NH_ * NP_ * D_ * 2);
  _Float16* doutW16 = ( _Float16*)palloc(D_ * D_ * 2);
  _Float16* W1_16 = ( _Float16*)palloc((size_t)DFF_ * D_ * 2);
  _Float16* W2_16 = ( _Float16*)palloc((size_t)D_ * DFF_ * 2);
  float* x = (float*)palloc(nBQD * 4);
  _Float16* x16 = (_Float16*)palloc(nBQD * 2);
  float* x2 = (float*)palloc(nBQD * 4);
  _Float16* x216 = (_Float16*)palloc(nBQD * 2);

  const size_t arena0 = po;
  size_t ao = arena0;
  auto aalloc = [&](size_t bytes) -> void* {
    void* p = base + ao; ao = (ao + bytes + 255) & ~(size_t)255; return p;
  };

  auto cvt = [&](const float* in, _Float16* out, size_t n) {
    int blocks = (int)((n + 255) / 256);
    cvt_f32_to_f16_kernel<<<blocks, 256, 0, stream>>>(in, out, n);
  };
  auto gemm = [&](const _Float16* A, const _Float16* W, const float* bias,
                  float* oF, _Float16* oH, int M, int N, int K, int act) {
    int mt = M >> 4;
    if (N % 64 == 0) {
      int nt = N / 64;
      int blocks = (mt * nt + 7) / 8;
      wmma_gemm_kernel<4><<<blocks, 256, 0, stream>>>(A, W, bias, oF, oH, M, N, K, act);
    } else if (N == 48) {
      int blocks = (mt + 7) / 8;
      wmma_gemm_kernel<3><<<blocks, 256, 0, stream>>>(A, W, bias, oF, oH, M, N, K, act);
    } else {  // N == 24
      int blocks = (mt + 7) / 8;
      wmma_gemm_kernel<2><<<blocks, 256, 0, stream>>>(A, W, bias, oF, oH, M, N, K, act);
    }
  };

  // weight conversions
  cvt(Wq, Wq16, (size_t)D_ * D_);
  cvt(Wk, Wk16, (size_t)D_ * D_);
  cvt(Wv, Wv16, (size_t)D_ * D_);
  cvt(Wo, Wo16, (size_t)D_ * D_);
  cvt(dvW, dvW16, (size_t)D_ * D_);
  cvt(doffW, doffW16, (size_t)(NH_ * NP_ * 2) * D_);
  cvt(dawW, dawW16, (size_t)(NH_ * NP_) * D_);
  cvt(doutW, doutW16, (size_t)D_ * D_);
  cvt(W1, W1_16, (size_t)DFF_ * D_);
  cvt(W2, W2_16, (size_t)D_ * DFF_);

  // ---------- phase A: self-attention ----------
  _Float16* qint16 = (_Float16*)aalloc(nBQD * 2);
  _Float16* query16 = (_Float16*)aalloc(nBQD * 2);
  _Float16* qh16 = (_Float16*)aalloc(nBQD * 2);
  _Float16* kh16 = (_Float16*)aalloc(nBQD * 2);
  _Float16* vh16 = (_Float16*)aalloc(nBQD * 2);
  _Float16* vT16 = (_Float16*)aalloc(nBQD * 2);
  _Float16* attn16 = (_Float16*)aalloc(nBQD * 2);
  float* oproj = (float*)aalloc(nBQD * 4);

  {
    int blocks = (int)((nBQD + 255) / 256);
    add_cvt_f16_kernel<<<blocks, 256, 0, stream>>>(query, pos, qint16, nBQD);
  }
  cvt(query, query16, nBQD);
  gemm(qint16, Wq16, bq, nullptr, qh16, (int)BQ, D_, D_, /*scale 1/sqrt(dh)*/2);
  gemm(qint16, Wk16, bk, nullptr, kh16, (int)BQ, D_, D_, 0);
  gemm(query16, Wv16, bv, nullptr, vh16, (int)BQ, D_, D_, 0);
  {
    int blocks = (int)((nBQD + 255) / 256);
    vtrans_kernel<<<blocks, 256, 0, stream>>>(vh16, vT16);
  }
  {
    int waves = B_ * NH_ * (Q_ / 16);   // 3072
    flash_attn_kernel<<<(waves + 7) / 8, 256, 0, stream>>>(qh16, kh16, vT16, attn16);
  }
  gemm(attn16, Wo16, bo, oproj, nullptr, (int)BQ, D_, D_, 0);
  ln_kernel<<<(int)((BQ + 7) / 8), 256, 0, stream>>>(query, oproj, ln1g, ln1b, x, x16, (int)BQ);

  // ---------- phase B: deformable cross-attention (arena reused) ----------
  ao = arena0;
  _Float16* src16 = (_Float16*)aalloc(nBSD * 2);
  float* vproj = (float*)aalloc(nBSD * 4);
  float* offb = (float*)aalloc(BQ * NH_ * NP_ * 2 * 4);
  float* awl = (float*)aalloc(BQ * NH_ * NP_ * 4);
  _Float16* cross16 = (_Float16*)aalloc(nBQD * 2);
  float* crossp = (float*)aalloc(nBQD * 4);

  cvt(src, src16, nBSD);
  gemm(src16, dvW16, dvb, vproj, nullptr, (int)BS, D_, D_, 0);
  gemm(x16, doffW16, doffb, offb, nullptr, (int)BQ, NH_ * NP_ * 2, D_, 0);
  gemm(x16, dawW16, dawb, awl, nullptr, (int)BQ, NH_ * NP_, D_, 0);
  {
    int total = B_ * Q_ * NH_ * DH_;
    deform_kernel<<<(total + 255) / 256, 256, 0, stream>>>(vproj, offb, awl, refp, cross16);
  }
  gemm(cross16, doutW16, doutb, crossp, nullptr, (int)BQ, D_, D_, 0);
  ln_kernel<<<(int)((BQ + 7) / 8), 256, 0, stream>>>(x, crossp, ln2g, ln2b, x2, x216, (int)BQ);

  // ---------- phase C: FFN (arena reused) ----------
  ao = arena0;
  _Float16* ffh16 = (_Float16*)aalloc(BQ * DFF_ * 2);
  float* ff2 = (float*)aalloc(nBQD * 4);

  gemm(x216, W1_16, b1, nullptr, ffh16, (int)BQ, DFF_, D_, /*gelu*/1);
  gemm(ffh16, W2_16, b2, ff2, nullptr, (int)BQ, D_, DFF_, 0);
  ln_kernel<<<(int)((BQ + 7) / 8), 256, 0, stream>>>(x2, ff2, ln3g, ln3b, outp, nullptr, (int)BQ);
}
